// DotProductAttention_2568390443100
// MI455X (gfx1250) — matrix operand order repaired and checked by reading, not compiled, not run on hardware
//
#include <hip/hip_runtime.h>
#include <math.h>
#include <stdint.h>

typedef __attribute__((ext_vector_type(16))) _Float16 v16h;
typedef __attribute__((ext_vector_type(8)))  float    v8f;

#define WMMA_F16(a, b, c) \
  __builtin_amdgcn_wmma_f32_16x16x32_f16(false, (a), false, (b), (short)0, (c), false, false)

// ---------------------------------------------------------------------------
// Fragment gathers from LDS (wave32, layouts per CDNA5 ISA 7.12.2)
// A: 16x32 f16 tile, row-major [16][stride]; lane m=L%16, half=L/16,
//    VGPR v holds K = koff + 8*half + 2*(v&3) + 16*(v>>2), +1.
// B: 32x16 f16 tile, stored as [n][k] row-major [16][stride]; lane n=L%16,
//    khalf=L/16, VGPR v holds K = koff + 8*khalf + 2*(v&3) + 16*(v>>2), +1.
// ---------------------------------------------------------------------------
static __device__ __forceinline__ v16h frag_A(const _Float16* __restrict__ base,
                                              int stride, int koff) {
  const int lane = threadIdx.x & 31;
  const int m = lane & 15, half = lane >> 4;
  const _Float16* row = base + m * stride + koff + 8 * half;
  v16h a;
#pragma unroll
  for (int v = 0; v < 8; ++v) {
    const int k = 2 * (v & 3) + 16 * (v >> 2);
    a[2 * v]     = row[k];
    a[2 * v + 1] = row[k + 1];
  }
  return a;
}

static __device__ __forceinline__ v16h frag_B(const _Float16* __restrict__ base,
                                              int stride, int koff) {
  const int lane = threadIdx.x & 31;
  const int n = lane & 15, khalf = lane >> 4;
  const _Float16* row = base + n * stride + koff + 8 * khalf;
  v16h b;
#pragma unroll
  for (int v = 0; v < 8; ++v) {
    const int k = 2 * (v & 3) + 16 * (v >> 2);
    b[2 * v]     = row[k];
    b[2 * v + 1] = row[k + 1];
  }
  return b;
}

// CDNA5 async global->LDS copy (ASYNCcnt-tracked), GV addressing.
// Low 32 bits of a generic pointer to LDS are the LDS byte offset.
static __device__ __forceinline__ void async_ld_lds_b128(uint32_t lds_off,
                                                         const void* gptr) {
  asm volatile("global_load_async_to_lds_b128 %0, %1, off"
               :: "v"(lds_off), "v"((uint64_t)(uintptr_t)gptr)
               : "memory");
}
static __device__ __forceinline__ void wait_async0() {
  asm volatile("s_wait_asynccnt 0" ::: "memory");
}

// ---------------------------------------------------------------------------
// Kernel 1: QKV projection. Y[M,1024] = f16((X[M,1024] @ W[1024,1024]) + bias)
// (optionally * per-dim-scale for Q). Block tile 128x64, 8 waves; each wave
// owns a 32x32 output tile -> 4 WMMA per k-step with 2 A + 2 B fragments.
// ---------------------------------------------------------------------------
__global__ __launch_bounds__(256) void qkv_proj_kernel(
    const float* __restrict__ X, const float* __restrict__ W,
    const float* __restrict__ bias, const float* __restrict__ pds,
    _Float16* __restrict__ Y) {
  __shared__ __align__(16) _Float16 As[128 * 36];
  __shared__ __align__(16) _Float16 Bt[64 * 36];
  const int tid = threadIdx.x;
  const int c0 = blockIdx.x * 64, m0 = blockIdx.y * 128;
  const int w = tid >> 5, lane = tid & 31;
  const int mh = w >> 1, nh = w & 1;
  const int lane16 = lane & 15, half = lane >> 4;

  v8f acc[4];
#pragma unroll
  for (int i = 0; i < 4; ++i)
#pragma unroll
    for (int r = 0; r < 8; ++r) acc[i][r] = 0.0f;

  const int arow = tid >> 1, akseg = (tid & 1) * 16;
  const int bk = tid >> 3, bcseg = (tid & 7) * 8;

  for (int k0 = 0; k0 < 1024; k0 += 32) {
    __syncthreads();
    const float* xsrc = X + (size_t)(m0 + arow) * 1024 + k0 + akseg;
#pragma unroll
    for (int j = 0; j < 16; ++j) As[arow * 36 + akseg + j] = (_Float16)xsrc[j];
    const float* wsrc = W + (size_t)(k0 + bk) * 1024 + c0 + bcseg;
#pragma unroll
    for (int j = 0; j < 8; ++j) Bt[(bcseg + j) * 36 + bk] = (_Float16)wsrc[j];
    if (k0 + 32 < 1024) {
      __builtin_prefetch(xsrc + 32, 0, 1);
      __builtin_prefetch(wsrc + 32 * 1024, 0, 1);
    }
    __syncthreads();
    const v16h bf0 = frag_B(Bt + (nh * 32) * 36, 36, 0);
    const v16h bf1 = frag_B(Bt + (nh * 32 + 16) * 36, 36, 0);
    const v16h af0 = frag_A(As + (mh * 32) * 36, 36, 0);
    const v16h af1 = frag_A(As + (mh * 32 + 16) * 36, 36, 0);
    acc[0] = WMMA_F16(af0, bf0, acc[0]);
    acc[1] = WMMA_F16(af0, bf1, acc[1]);
    acc[2] = WMMA_F16(af1, bf0, acc[2]);
    acc[3] = WMMA_F16(af1, bf1, acc[3]);
  }

  float sc[2], bv[2];
  int cN[2];
#pragma unroll
  for (int ni = 0; ni < 2; ++ni) {
    cN[ni] = c0 + nh * 32 + ni * 16 + lane16;
    sc[ni] = 1.0f;
    if (pds) {
      const float x = pds[cN[ni] & 63];
      const float sp = (x > 20.0f) ? x : log1pf(__expf(x));  // softplus
      sc[ni] = 0.180336880131183f * sp;  // (1/ln2)/sqrt(64)
    }
    bv[ni] = bias[cN[ni]];
  }
#pragma unroll
  for (int mi = 0; mi < 2; ++mi)
#pragma unroll
    for (int ni = 0; ni < 2; ++ni)
#pragma unroll
      for (int r = 0; r < 8; ++r) {
        const int row = m0 + mh * 32 + mi * 16 + r + 8 * half;
        Y[(size_t)row * 1024 + cN[ni]] =
            (_Float16)((acc[mi * 2 + ni][r] + bv[ni]) * sc[ni]);
      }
}

// ---------------------------------------------------------------------------
// Kernel 2: attention for one (b, n, 16-row Q tile). Flash-style online
// softmax; raw capped logits streamed to probs output; PV accumulated in f32.
// K chunks staged with CDNA5 async global->LDS copies.
// ---------------------------------------------------------------------------
__global__ __launch_bounds__(256) void attn_kernel(
    const _Float16* __restrict__ Q, const _Float16* __restrict__ K,
    const _Float16* __restrict__ V, _Float16* __restrict__ Enc,
    float* __restrict__ Probs, float* __restrict__ Mws,
    float* __restrict__ RLws) {
  __shared__ __align__(16) _Float16 Qs[16 * 72];
  __shared__ __align__(16) _Float16 KVs[256 * 72];  // K [s][h]/72; V [h][s]/264
  __shared__ __align__(16) _Float16 Ps[8][16 * 36];
  __shared__ float Of[16 * 64];
  __shared__ float Ms[8 * 16], Ls[8 * 16], Lf[16];

  const int tid = threadIdx.x, w = tid >> 5, lane = tid & 31;
  const int lane16 = lane & 15, half = lane >> 4;
  const int t0 = blockIdx.x * 16, n = blockIdx.y, b = blockIdx.z;
  const int bn = b * 16 + n;
  const uint32_t kvs_lds = (uint32_t)(uintptr_t)(&KVs[0]);

#pragma unroll
  for (int i = 0; i < 4; ++i) {
    const int idx = tid * 4 + i;
    const int tl = idx >> 6, h = idx & 63;
    Qs[tl * 72 + h] = Q[(size_t)(b * 2048 + t0 + tl) * 1024 + n * 64 + h];
  }
  __syncthreads();
  const v16h aq0 = frag_A(Qs, 72, 0);
  const v16h aq1 = frag_A(Qs, 72, 32);

  float m_r[8], l_r[8];
  v8f o_acc[4];
#pragma unroll
  for (int r = 0; r < 8; ++r) { m_r[r] = -1e30f; l_r[r] = 0.0f; }
#pragma unroll
  for (int j = 0; j < 4; ++j)
#pragma unroll
    for (int r = 0; r < 8; ++r) o_acc[j][r] = 0.0f;

  _Float16* Vt = KVs;  // [64][264] view

  for (int sblk = 0; sblk < 8; ++sblk) {
    const int s_base = sblk * 256;
    __syncthreads();
    // stage K chunk [256][64]f16 -> LDS [s][h] (stride 72) via async DMA:
    // each thread copies 4 x 16B (one 8-f16 row segment at h = hseg per pass).
#pragma unroll
    for (int p = 0; p < 4; ++p) {
      const int row = p * 64 + (tid >> 2), hseg = (tid & 3) * 16;
      const _Float16* src =
          K + (size_t)(b * 2048 + s_base + row) * 1024 + n * 64 + hseg;
      async_ld_lds_b128(kvs_lds + (uint32_t)(row * 144 + hseg * 2), src);
    }
    wait_async0();
    __syncthreads();

    float xt[2][8];
#pragma unroll
    for (int tile = 0; tile < 2; ++tile) {
      const int s_off = 32 * w + 16 * tile;
      const v16h b0 = frag_B(KVs + s_off * 72, 72, 0);
      const v16h b1 = frag_B(KVs + s_off * 72, 72, 32);
      v8f x;
#pragma unroll
      for (int r = 0; r < 8; ++r) x[r] = 0.0f;
      x = WMMA_F16(aq0, b0, x);
      x = WMMA_F16(aq1, b1, x);
#pragma unroll
      for (int r = 0; r < 8; ++r) {
        const float xc = 50.0f * tanhf(x[r] * 0.02f);  // soft cap
        xt[tile][r] = xc;
        const size_t rowg = (size_t)bn * 2048 + (t0 + r + 8 * half);
        Probs[rowg * 2048 + (s_base + s_off + lane16)] = xc;
      }
    }
    // online softmax over this wave's 32 columns
#pragma unroll
    for (int r = 0; r < 8; ++r) {
      float vm = fmaxf(xt[0][r], xt[1][r]);
      for (int o = 1; o < 16; o <<= 1) vm = fmaxf(vm, __shfl_xor(vm, o, 16));
      const float mnew = fmaxf(m_r[r], vm);
      const float alpha = __expf(m_r[r] - mnew);
      const float p0 = __expf(xt[0][r] - mnew);
      const float p1 = __expf(xt[1][r] - mnew);
      float rs = p0 + p1;
      for (int o = 1; o < 16; o <<= 1) rs += __shfl_xor(rs, o, 16);
      l_r[r] = l_r[r] * alpha + rs;
      m_r[r] = mnew;
#pragma unroll
      for (int j = 0; j < 4; ++j) o_acc[j][r] *= alpha;
      Ps[w][(r + 8 * half) * 36 + lane16] = (_Float16)p0;
      Ps[w][(r + 8 * half) * 36 + 16 + lane16] = (_Float16)p1;
    }
    __syncthreads();
    // stage V chunk transposed -> LDS [h][s] (stride 264)
#pragma unroll
    for (int p = 0; p < 4; ++p) {
      const int row = p * 64 + (tid >> 2), hseg = (tid & 3) * 16;
      const _Float16* src =
          V + (size_t)(b * 2048 + s_base + row) * 1024 + n * 64 + hseg;
#pragma unroll
      for (int j = 0; j < 16; ++j) Vt[(hseg + j) * 264 + row] = src[j];
    }
    __syncthreads();
    const v16h ap = frag_A(Ps[w], 36, 0);
#pragma unroll
    for (int j = 0; j < 4; ++j) {
      const v16h bv = frag_B(Vt + (16 * j) * 264 + 32 * w, 264, 0);
      o_acc[j] = WMMA_F16(ap, bv, o_acc[j]);
    }
  }

  // ---- cross-wave merge of (m, l, O) ----
  if (lane16 == 0) {
#pragma unroll
    for (int r = 0; r < 8; ++r) {
      Ms[w * 16 + r + 8 * half] = m_r[r];
      Ls[w * 16 + r + 8 * half] = l_r[r];
    }
  }
#pragma unroll
  for (int i = 0; i < 4; ++i) Of[tid * 4 + i] = 0.0f;
  __syncthreads();
#pragma unroll
  for (int r = 0; r < 8; ++r) {
    const int t = r + 8 * half;
    float mf = -1e30f;
    for (int ww = 0; ww < 8; ++ww) mf = fmaxf(mf, Ms[ww * 16 + t]);
    float lf = 0.0f;
    for (int ww = 0; ww < 8; ++ww)
      lf += Ls[ww * 16 + t] * __expf(Ms[ww * 16 + t] - mf);
    const float alpha = __expf(m_r[r] - mf);
#pragma unroll
    for (int j = 0; j < 4; ++j)
      atomicAdd(&Of[t * 64 + 16 * j + lane16], o_acc[j][r] * alpha);
    if (w == 0 && lane16 == 0) {
      Lf[t] = lf;
      const size_t rg = (size_t)bn * 2048 + t0 + t;
      Mws[rg] = mf;
      RLws[rg] = 1.0f / lf;
    }
  }
  __syncthreads();
#pragma unroll
  for (int i = 0; i < 4; ++i) {
    const int idx = tid * 4 + i;
    const int tl = idx >> 6, h = idx & 63;
    Enc[(size_t)(b * 2048 + t0 + tl) * 1024 + n * 64 + h] =
        (_Float16)(Of[idx] / Lf[tl]);
  }
}

// ---------------------------------------------------------------------------
// Kernel 3: normalize the stored capped logits into softmax probabilities.
// ---------------------------------------------------------------------------
__global__ __launch_bounds__(256) void norm_probs_kernel(
    float* __restrict__ P, const float* __restrict__ M,
    const float* __restrict__ RL) {
  const int row = blockIdx.x;
  const float m = M[row], rl = RL[row];
  float4* p = (float4*)(P + (size_t)row * 2048);
  for (int i = threadIdx.x; i < 512; i += 256) {
    float4 v = p[i];
    v.x = __expf(v.x - m) * rl;
    v.y = __expf(v.y - m) * rl;
    v.z = __expf(v.z - m) * rl;
    v.w = __expf(v.w - m) * rl;
    p[i] = v;
  }
}

// ---------------------------------------------------------------------------
// Kernel 4: output projection. out[M,1024] = enc[M,1024] @ w_post^T + b_post
// (w_post is [D][N*H], i.e. already in [n][k] order for B fragments).
// Block tile 128x64, 32x32 per wave, 4 WMMA per k-step.
// ---------------------------------------------------------------------------
__global__ __launch_bounds__(256) void post_proj_kernel(
    const _Float16* __restrict__ A, const float* __restrict__ Wp,
    const float* __restrict__ bias, float* __restrict__ Out) {
  __shared__ __align__(16) _Float16 As[128 * 36];
  __shared__ __align__(16) _Float16 Bt[64 * 36];
  const int tid = threadIdx.x;
  const int c0 = blockIdx.x * 64, m0 = blockIdx.y * 128;
  const int w = tid >> 5, lane = tid & 31;
  const int mh = w >> 1, nh = w & 1;
  const int lane16 = lane & 15, half = lane >> 4;

  v8f acc[4];
#pragma unroll
  for (int i = 0; i < 4; ++i)
#pragma unroll
    for (int r = 0; r < 8; ++r) acc[i][r] = 0.0f;

  const int arow = tid >> 1, akseg = (tid & 1) * 16;
  const int brow = tid >> 2, bkseg = (tid & 3) * 8;

  for (int k0 = 0; k0 < 1024; k0 += 32) {
    __syncthreads();
    const _Float16* asrc = A + (size_t)(m0 + arow) * 1024 + k0 + akseg;
#pragma unroll
    for (int j = 0; j < 16; ++j) As[arow * 36 + akseg + j] = asrc[j];
    const float* wsrc = Wp + (size_t)(c0 + brow) * 1024 + k0 + bkseg;
#pragma unroll
    for (int j = 0; j < 8; ++j) Bt[brow * 36 + bkseg + j] = (_Float16)wsrc[j];
    if (k0 + 32 < 1024) {
      __builtin_prefetch(asrc + 32, 0, 1);
      __builtin_prefetch(wsrc + 32, 0, 1);
    }
    __syncthreads();
    const v16h bf0 = frag_B(Bt + (nh * 32) * 36, 36, 0);
    const v16h bf1 = frag_B(Bt + (nh * 32 + 16) * 36, 36, 0);
    const v16h af0 = frag_A(As + (mh * 32) * 36, 36, 0);
    const v16h af1 = frag_A(As + (mh * 32 + 16) * 36, 36, 0);
    acc[0] = WMMA_F16(af0, bf0, acc[0]);
    acc[1] = WMMA_F16(af0, bf1, acc[1]);
    acc[2] = WMMA_F16(af1, bf0, acc[2]);
    acc[3] = WMMA_F16(af1, bf1, acc[3]);
  }

#pragma unroll
  for (int ni = 0; ni < 2; ++ni) {
    const int c = c0 + nh * 32 + ni * 16 + lane16;
    const float bv = bias[c];
#pragma unroll
    for (int mi = 0; mi < 2; ++mi)
#pragma unroll
      for (int r = 0; r < 8; ++r) {
        const int row = m0 + mh * 32 + mi * 16 + r + 8 * half;
        Out[(size_t)row * 1024 + c] = acc[mi * 2 + ni][r] + bv;
      }
  }
}

// ---------------------------------------------------------------------------
extern "C" void kernel_launch(void* const* d_in, const int* in_sizes, int n_in,
                              void* d_out, int out_size, void* d_ws,
                              size_t ws_size, hipStream_t stream) {
  const float* qv = (const float*)d_in[0];
  const float* kv = (const float*)d_in[1];
  const float* vv = (const float*)d_in[2];
  const float* wq = (const float*)d_in[3];
  const float* bq = (const float*)d_in[4];
  const float* wk = (const float*)d_in[5];
  const float* bk = (const float*)d_in[6];
  const float* wv = (const float*)d_in[7];
  const float* bv = (const float*)d_in[8];
  const float* wp = (const float*)d_in[9];
  const float* bp = (const float*)d_in[10];
  const float* pds = (const float*)d_in[11];

  _Float16* qws = (_Float16*)d_ws;          // 4096x1024 f16
  _Float16* kws = qws + 4096 * 1024;
  _Float16* vws = kws + 4096 * 1024;
  _Float16* ews = vws + 4096 * 1024;        // encoded
  float* mws = (float*)(ews + 4096 * 1024); // [B*N*T] row max
  float* rlws = mws + 65536;                // [B*N*T] 1/l

  float* out = (float*)d_out;               // [2,2048,1024]
  float* probs = out + 4194304;             // [2,16,2048,2048]

  const dim3 bl(256), gp(16, 32);
  qkv_proj_kernel<<<gp, bl, 0, stream>>>(qv, wq, bq, pds, qws);
  qkv_proj_kernel<<<gp, bl, 0, stream>>>(kv, wk, bk, (const float*)nullptr, kws);
  qkv_proj_kernel<<<gp, bl, 0, stream>>>(vv, wv, bv, (const float*)nullptr, vws);
  attn_kernel<<<dim3(128, 16, 2), bl, 0, stream>>>(qws, kws, vws, ews, probs,
                                                   mws, rlws);
  norm_probs_kernel<<<dim3(65536), bl, 0, stream>>>(probs, mws, rlws);
  post_proj_kernel<<<gp, bl, 0, stream>>>(ews, wp, bp, out);
}
